// ExllamaLinear_54657753808996
// MI455X (gfx1250) — compile-verified, matrix operand order checked by reading, not run on hardware
//
#include <hip/hip_runtime.h>

typedef _Float16 h2   __attribute__((ext_vector_type(2)));
typedef _Float16 v8h  __attribute__((ext_vector_type(8)));
typedef _Float16 v16h __attribute__((ext_vector_type(16)));
typedef float    v8f  __attribute__((ext_vector_type(8)));

#define TOKENS  32
#define KDIM    8192
#define NDIM    28672
#define GS      128
#define NGROUPS (KDIM / GS)

// One wave computes a 32(M) x 16(N) output tile across the full K dimension.
// 4 waves / block -> 64 columns per block; grid = 28672/64 = 448 blocks
// (more, smaller blocks than before so the workgroup scheduler can span every
//  WGP and saturate the 23.3 TB/s HBM floor; traffic is unchanged).
__global__ __launch_bounds__(128)
void exllama_q4_wmma(const _Float16* __restrict__ x,
                     const int*      __restrict__ qweight,
                     const int*      __restrict__ qzeros,
                     const _Float16* __restrict__ scales,
                     const _Float16* __restrict__ bias,
                     float*          __restrict__ out)
{
    const int lane     = threadIdx.x & 31;
    const int wave     = threadIdx.x >> 5;
    const int half_sel = lane >> 4;      // 0: K 0-7 / 16-23   1: K 8-15 / 24-31
    const int nlane    = lane & 15;
    const int n_base   = blockIdx.x * 64 + wave * 16;
    const int n        = n_base + nlane; // this lane's B column (also its C/D column)

    v8f c0 = {};                         // rows  0..15
    v8f c1 = {};                         // rows 16..31

    const _Float16* xrow0 = x + (size_t)nlane        * KDIM;
    const _Float16* xrow1 = x + (size_t)(nlane + 16) * KDIM;

    for (int g = 0; g < NGROUPS; ++g) {
        // Per-group dequant constants for this lane's column:
        //   w = s*(q - (z+1)) = s*(1024+q) - s*(1025+z)   with magic f16 0x6400 = 1024.
        const _Float16 s  = scales[(size_t)g * NDIM + n];
        const unsigned zw = (unsigned)qzeros[(size_t)g * (NDIM / 8) + (n >> 3)];
        const int z       = (zw >> (4 * (n & 7))) & 15;
        const _Float16 off = (_Float16)(-(float)s * (float)(1025 + z));
        const h2 s2 = { s,   s   };
        const h2 o2 = { off, off };

        const int kg = g * GS;
        #pragma unroll
        for (int kk = 0; kk < GS; kk += 32) {
            const int kb = kg + kk;
            const int r0 = (kb >> 3) + half_sel;   // packed row for this lane's low K-octet
            const unsigned w0 = (unsigned)qweight[(size_t)r0       * NDIM + n];
            const unsigned w1 = (unsigned)qweight[(size_t)(r0 + 2) * NDIM + n];
            if (kb + GS < KDIM)                    // pull next group's word into near cache
                __builtin_prefetch(&qweight[(size_t)(r0 + 16) * NDIM + n], 0, 3);

            // Dequantize 16 nibbles -> B fragment (K ascending within lane).
            v16h b;
            #pragma unroll
            for (int p = 0; p < 4; ++p) {
                unsigned u0 = w0 >> (8 * p);
                unsigned t0 = (u0 & 0xFu) | ((u0 & 0xF0u) << 12) | 0x64006400u;
                h2 v0 = __builtin_bit_cast(h2, t0) * s2 + o2;   // v_pk_fma_f16
                b[2 * p + 0] = v0[0];
                b[2 * p + 1] = v0[1];

                unsigned u1 = w1 >> (8 * p);
                unsigned t1 = (u1 & 0xFu) | ((u1 & 0xF0u) << 12) | 0x64006400u;
                h2 v1 = __builtin_bit_cast(h2, t1) * s2 + o2;
                b[8 + 2 * p + 0] = v1[0];
                b[8 + 2 * p + 1] = v1[1];
            }

            // A fragments: two aligned b128 loads per 16-row tile (L2-resident x).
            const int ka = kb + half_sel * 8;
            v8h a0lo = *(const v8h*)(xrow0 + ka);
            v8h a0hi = *(const v8h*)(xrow0 + ka + 16);
            v8h a1lo = *(const v8h*)(xrow1 + ka);
            v8h a1hi = *(const v8h*)(xrow1 + ka + 16);
            v16h a0 = __builtin_shufflevector(a0lo, a0hi,
                        0,1,2,3,4,5,6,7,8,9,10,11,12,13,14,15);
            v16h a1 = __builtin_shufflevector(a1lo, a1hi,
                        0,1,2,3,4,5,6,7,8,9,10,11,12,13,14,15);

            c0 = __builtin_amdgcn_wmma_f32_16x16x32_f16(false, a0, false, b,
                                                        (short)0, c0, false, false);
            c1 = __builtin_amdgcn_wmma_f32_16x16x32_f16(false, a1, false, b,
                                                        (short)0, c1, false, false);
        }
    }

    // Epilogue: C/D layout -> VGPR r holds row r (lanes 0-15) / row r+8 (lanes 16-31).
    const float bf = (float)bias[n];
    #pragma unroll
    for (int r = 0; r < 8; ++r) {
        const int row = r + 8 * half_sel;
        out[(size_t)row        * NDIM + n] = c0[r] + bf;
        out[(size_t)(row + 16) * NDIM + n] = c1[r] + bf;
    }
}

extern "C" void kernel_launch(void* const* d_in, const int* in_sizes, int n_in,
                              void* d_out, int out_size, void* d_ws, size_t ws_size,
                              hipStream_t stream) {
    const _Float16* x       = (const _Float16*)d_in[0];
    const int*      qweight = (const int*)d_in[1];
    const int*      qzeros  = (const int*)d_in[2];
    const _Float16* scales  = (const _Float16*)d_in[3];
    const _Float16* bias    = (const _Float16*)d_in[4];
    float*          out     = (float*)d_out;

    dim3 grid(NDIM / 64);    // 448 workgroups
    dim3 block(128);         // 4 wave32 waves
    hipLaunchKernelGGL(exllama_q4_wmma, grid, block, 0, stream,
                       x, qweight, qzeros, scales, bias, out);
}
